// BEVGNNModelLoaded_81492709474596
// MI455X (gfx1250) — matrix operand order repaired and checked by reading, not compiled, not run on hardware
//
#include <hip/hip_runtime.h>

typedef __attribute__((ext_vector_type(16))) __bf16 v16bf;
typedef __attribute__((ext_vector_type(8)))  float  v8f;

#define BB 32
#define NN 8
#define CIN 3
#define HH 224
#define WW 224
#define PATCH 32
#define CC 384
#define SS 49
#define HM 512
#define CO 256
#define ROWS (BB * NN * SS)      // 12544
#define KPATCH (CIN * PATCH * PATCH) // 3072

// ---------------------------------------------------------------------------
// Zero the whole output buffer (edge_out + mask get overwritten later;
// node_preds region must be zeros every call).
__global__ void k_zero(float* out, int n) {
    int i = blockIdx.x * blockDim.x + threadIdx.x;
    if (i < n) out[i] = 0.0f;
}

// f32 -> bf16 elementwise (W_patch is already in [c][k] = Bt layout).
__global__ void k_cvt_bf16(const float* __restrict__ src, __bf16* __restrict__ dst, int n) {
    int i = blockIdx.x * blockDim.x + threadIdx.x;
    if (i < n) dst[i] = (__bf16)src[i];
}

// W1 (C x HM, row-major) -> Wt (HM x C, bf16)  [B-transposed layout for WMMA]
__global__ void k_transpose_w1(const float* __restrict__ W, __bf16* __restrict__ Wt) {
    int i = blockIdx.x * blockDim.x + threadIdx.x;
    if (i < HM * CC) {
        int h = i / CC, c = i % CC;
        Wt[i] = (__bf16)W[c * HM + h];
    }
}

__device__ inline float head_col(const float* Wp, const float* Wpv,
                                 const float* Wr, const float* Wrv, int o, int j) {
    switch (j) {
        case 0: return Wp [o * 2 + 0];
        case 1: return Wp [o * 2 + 1];
        case 2: return Wpv[o * 2 + 0];
        case 3: return Wpv[o * 2 + 1];
        case 4: return Wr [o * 2 + 0];
        case 5: return Wr [o * 2 + 1];
        default: return Wrv[o];
    }
}

// Fold the 4 head projections through W2: W27[h][j] = sum_o W2[h,o]*Whead[o,j]
// const7[j] = b2 @ Whead[:,j] + bhead[j]
__global__ void k_fold_heads(const float* __restrict__ W2, const float* __restrict__ b2,
                             const float* __restrict__ Wp,  const float* __restrict__ bp,
                             const float* __restrict__ Wpv, const float* __restrict__ bpv,
                             const float* __restrict__ Wr,  const float* __restrict__ br,
                             const float* __restrict__ Wrv, const float* __restrict__ brv,
                             float* __restrict__ W27, float* __restrict__ const7) {
    int h = blockIdx.x * blockDim.x + threadIdx.x;
    if (h < HM) {
        float s[7] = {0.f, 0.f, 0.f, 0.f, 0.f, 0.f, 0.f};
        for (int o = 0; o < CO; ++o) {
            float w = W2[h * CO + o];
            s[0] += w * Wp [o * 2 + 0];
            s[1] += w * Wp [o * 2 + 1];
            s[2] += w * Wpv[o * 2 + 0];
            s[3] += w * Wpv[o * 2 + 1];
            s[4] += w * Wr [o * 2 + 0];
            s[5] += w * Wr [o * 2 + 1];
            s[6] += w * Wrv[o];
        }
        #pragma unroll
        for (int j = 0; j < 7; ++j) W27[h * 8 + j] = s[j];
        W27[h * 8 + 7] = 0.f;
    }
    if (h < 7) {
        float c = 0.f;
        for (int o = 0; o < CO; ++o) c += b2[o] * head_col(Wp, Wpv, Wr, Wrv, o, h);
        float bias = (h < 2) ? bp[h] : (h < 4) ? bpv[h - 2] : (h < 6) ? br[h - 4] : brv[0];
        const7[h] = c + bias;
    }
}

// Gather img (B*N, CIN, H, W) f32 into bf16 patch matrix Apat[ROWS][KPATCH].
__global__ void k_patchify(const float* __restrict__ img, __bf16* __restrict__ Apat) {
    size_t idx = (size_t)blockIdx.x * blockDim.x + threadIdx.x;
    const size_t total = (size_t)BB * NN * CIN * HH * WW;
    if (idx >= total) return;
    int x = (int)(idx % WW);
    size_t t = idx / WW;
    int y  = (int)(t % HH); t /= HH;
    int ci = (int)(t % CIN);
    int bn = (int)(t / CIN);
    int px = x >> 5, dx = x & 31, py = y >> 5, dy = y & 31;
    int s  = py * 7 + px;
    size_t row = (size_t)bn * SS + s;
    size_t k   = (size_t)ci * 1024 + dy * 32 + dx;
    Apat[row * KPATCH + k] = (__bf16)img[idx];
}

// ---------------------------------------------------------------------------
// WMMA bf16 GEMM:  out[M x N] = A[M x K] (row-major bf16) * Bt[N x K]^T (bf16)
// Workgroup: 256 threads = 8 waves, tile 128(M) x 128(N); wave tile 32 x 64.
// K-step 32 via v_wmma_f32_16x16x32_bf16. All dims divide exactly.
template <int K, bool OUT_BF16>
__global__ void k_gemm_wmma(const __bf16* __restrict__ A, const __bf16* __restrict__ Bt,
                            const float* __restrict__ bias, int N,
                            __bf16* __restrict__ outB, float* __restrict__ outF) {
    const int tid   = threadIdx.x;
    const int lane  = tid & 31;
    const int wave  = tid >> 5;
    const int wm    = wave & 3;          // 4 waves along M
    const int wn    = wave >> 2;         // 2 waves along N
    const int baseM = blockIdx.x * 128 + wm * 32;
    const int baseN = blockIdx.y * 128 + wn * 64;
    const int laneN = lane & 15;
    const int hiH   = (lane >> 4) * 8;   // D-fragment M offset for upper half-wave
    const int selA  = (lane >> 4) * 8;   // A-fragment K sub-select
    const int selB  = (lane >> 4) * 8;   // B-fragment K sub-select

    union V16 { v16bf v; uint4 q[2]; };

    v8f acc[2][4] = {};

    for (int kk = 0; kk < K; kk += 32) {
        V16 af[2];
        #pragma unroll
        for (int mi = 0; mi < 2; ++mi) {
            const __bf16* ap = A + (size_t)(baseM + mi * 16 + laneN) * K + kk + selA;
            af[mi].q[0] = *(const uint4*)(ap);
            af[mi].q[1] = *(const uint4*)(ap + 16);
            __builtin_prefetch(ap + 32, 0, 0);
        }
        #pragma unroll
        for (int ni = 0; ni < 4; ++ni) {
            V16 bf;
            const __bf16* bp = Bt + (size_t)(baseN + ni * 16 + laneN) * K + kk + selB;
            bf.q[0] = *(const uint4*)(bp);
            bf.q[1] = *(const uint4*)(bp + 16);
            #pragma unroll
            for (int mi = 0; mi < 2; ++mi) {
                acc[mi][ni] = __builtin_amdgcn_wmma_f32_16x16x32_bf16(
                    /*neg_a=*/false, af[mi].v, /*neg_b=*/false, bf.v,
                    /*c_mod=*/(short)0, acc[mi][ni],
                    /*reuse_a=*/false, /*reuse_b=*/false);
            }
        }
    }

    #pragma unroll
    for (int mi = 0; mi < 2; ++mi) {
        #pragma unroll
        for (int ni = 0; ni < 4; ++ni) {
            int col = baseN + ni * 16 + laneN;
            float bv = bias ? bias[col] : 0.0f;
            #pragma unroll
            for (int r = 0; r < 8; ++r) {
                int row = baseM + mi * 16 + r + hiH;
                float v = acc[mi][ni][r] + bv;
                if constexpr (OUT_BF16) outB[(size_t)row * N + col] = (__bf16)v;
                else                    outF[(size_t)row * N + col] = v;
            }
        }
    }
}

// ---------------------------------------------------------------------------
__global__ void k_mask(const float* __restrict__ pos, float* __restrict__ maskf,
                       float* __restrict__ out_mask) {
    int idx = blockIdx.x * blockDim.x + threadIdx.x;
    if (idx >= BB * NN * NN) return;
    int b = idx >> 6, i = (idx >> 3) & 7, j = idx & 7;
    float dxv = pos[(b * NN + i) * 3 + 0] - pos[(b * NN + j) * 3 + 0];
    float dyv = pos[(b * NN + i) * 3 + 1] - pos[(b * NN + j) * 3 + 1];
    float d2 = dxv * dxv + dyv * dyv;
    float m = (d2 < 0.25f && i != j) ? 1.0f : 0.0f;
    maskf[idx]    = m;
    out_mask[idx] = m;
}

// One block per edge (b,i,j): sum_s sum_h relu(a+g) * W27 -> 7 outputs.
// b1 is pre-folded into a.  a,g stay L2-resident (51 MB << 192 MB L2).
__global__ void k_edge(const float* __restrict__ a, const float* __restrict__ g,
                       const float* __restrict__ W27, const float* __restrict__ const7,
                       const float* __restrict__ maskf, float* __restrict__ edge_out) {
    __shared__ float w27[HM * 8];       // 16 KB
    __shared__ float red[256][8];       // 8 KB
    int tid = threadIdx.x;
    int blk = blockIdx.x;               // b*64 + i*8 + j
    int b = blk >> 6, i = (blk >> 3) & 7, j = blk & 7;

    for (int idx = tid; idx < HM * 8; idx += 256) w27[idx] = W27[idx];
    __syncthreads();

    const float* ar = a + (size_t)(b * NN + i) * SS * HM;
    const float* gr = g + (size_t)(b * NN + j) * SS * HM;

    float acc[7] = {0.f, 0.f, 0.f, 0.f, 0.f, 0.f, 0.f};
    for (int e = tid; e < SS * HM; e += 256) {
        float v = ar[e] + gr[e];
        v = v > 0.f ? v : 0.f;
        const float* wrow = &w27[(e & (HM - 1)) * 8];
        #pragma unroll
        for (int jj = 0; jj < 7; ++jj) acc[jj] += v * wrow[jj];
    }
    #pragma unroll
    for (int jj = 0; jj < 7; ++jj) red[tid][jj] = acc[jj];
    __syncthreads();
    if (tid < 7) {
        float s = 0.f;
        for (int t = 0; t < 256; ++t) s += red[t][tid];
        edge_out[blk * 7 + tid] = (s * (1.0f / (float)SS) + const7[tid]) * maskf[blk];
    }
}

// ---------------------------------------------------------------------------
extern "C" void kernel_launch(void* const* d_in, const int* in_sizes, int n_in,
                              void* d_out, int out_size, void* d_ws, size_t ws_size,
                              hipStream_t stream) {
    const float* img     = (const float*)d_in[0];
    const float* pos     = (const float*)d_in[1];
    const float* W_patch = (const float*)d_in[3];
    const float* b_patch = (const float*)d_in[4];
    const float* W1a     = (const float*)d_in[5];
    const float* W1b     = (const float*)d_in[6];
    const float* b1      = (const float*)d_in[7];
    const float* W2      = (const float*)d_in[8];
    const float* b2      = (const float*)d_in[9];
    const float* Wp  = (const float*)d_in[10]; const float* bp  = (const float*)d_in[11];
    const float* Wpv = (const float*)d_in[12]; const float* bpv = (const float*)d_in[13];
    const float* Wr  = (const float*)d_in[14]; const float* br  = (const float*)d_in[15];
    const float* Wrv = (const float*)d_in[16]; const float* brv = (const float*)d_in[17];
    float* out = (float*)d_out;

    // Workspace carve-up (256B aligned).
    char*  ws  = (char*)d_ws;
    size_t off = 0;
    auto carve = [&](size_t bytes) -> void* {
        void* p = ws + off;
        off = (off + bytes + 255) & ~(size_t)255;
        return p;
    };
    __bf16* Apat   = (__bf16*)carve((size_t)ROWS * KPATCH * 2); // 77.1 MB
    __bf16* WpatT  = (__bf16*)carve((size_t)CC * KPATCH * 2);   //  2.4 MB
    __bf16* feats  = (__bf16*)carve((size_t)ROWS * CC * 2);     //  9.6 MB
    __bf16* W1aT   = (__bf16*)carve((size_t)HM * CC * 2);
    __bf16* W1bT   = (__bf16*)carve((size_t)HM * CC * 2);
    float*  aBuf   = (float*)carve((size_t)ROWS * HM * 4);      // 25.7 MB
    float*  gBuf   = (float*)carve((size_t)ROWS * HM * 4);      // 25.7 MB
    float*  W27    = (float*)carve((size_t)HM * 8 * 4);
    float*  const7 = (float*)carve(8 * 4);
    float*  maskf  = (float*)carve((size_t)BB * NN * NN * 4);

    const int edge_elems = BB * NN * NN * 7;  // 14336
    const int mask_elems = BB * NN * NN;      // 2048
    float* out_edge = out;
    float* out_mask = out + edge_elems;

    // 1. zero entire output (covers node_preds zeros)
    k_zero<<<(out_size + 255) / 256, 256, 0, stream>>>(out, out_size);

    // 2. weight prep
    k_cvt_bf16<<<(CC * KPATCH + 255) / 256, 256, 0, stream>>>(W_patch, WpatT, CC * KPATCH);
    k_transpose_w1<<<(HM * CC + 255) / 256, 256, 0, stream>>>(W1a, W1aT);
    k_transpose_w1<<<(HM * CC + 255) / 256, 256, 0, stream>>>(W1b, W1bT);
    k_fold_heads<<<2, 256, 0, stream>>>(W2, b2, Wp, bp, Wpv, bpv, Wr, br, Wrv, brv, W27, const7);

    // 3. img -> bf16 patch matrix
    {
        const size_t total = (size_t)BB * NN * CIN * HH * WW;
        k_patchify<<<(unsigned)((total + 255) / 256), 256, 0, stream>>>(img, Apat);
    }

    // 4. WMMA GEMMs
    //    feats = Apat(12544x3072) * WpatT^T (+b_patch)  -> bf16
    k_gemm_wmma<KPATCH, true><<<dim3(ROWS / 128, CC / 128), 256, 0, stream>>>(
        Apat, WpatT, b_patch, CC, feats, nullptr);
    //    a = feats * W1a (+b1 folded in), g = feats * W1b  -> f32
    k_gemm_wmma<CC, false><<<dim3(ROWS / 128, HM / 128), 256, 0, stream>>>(
        feats, W1aT, b1, HM, nullptr, aBuf);
    k_gemm_wmma<CC, false><<<dim3(ROWS / 128, HM / 128), 256, 0, stream>>>(
        feats, W1bT, nullptr, HM, nullptr, gBuf);

    // 5. mask + fused relu-edge reduction (x7 folded head weights)
    k_mask<<<(mask_elems + 255) / 256, 256, 0, stream>>>(pos, maskf, out_mask);
    k_edge<<<mask_elems, 256, 0, stream>>>(aBuf, gBuf, W27, const7, maskf, out_edge);
}
